// QuanvolutionGen112_65481071400242
// MI455X (gfx1250) — matrix operand order repaired and checked by reading, not compiled, not run on hardware
//
#include <hip/hip_runtime.h>
#include <hip/hip_bf16.h>
#include <stdint.h>

typedef __attribute__((ext_vector_type(16))) _Float16 v16h;
typedef __attribute__((ext_vector_type(8)))  _Float16 v8h;
typedef __attribute__((ext_vector_type(8)))  float    v8f;

// Problem constants
#define BATCH   8192
#define NPIX    196                 // 14*14
#define NPATCH  (BATCH * NPIX)      // 1,605,632 (divisible by 16)
#define KFC     3136                // 16*196
#define NCLS    10

// Workspace layout (all offsets 256B aligned)
//  qf_pad : [B][4][16][16] f16, zero border      -> 16,777,216 B
//  fused  : [B][3136] f16                        -> 51,380,224 B
//  fcw16  : [16][3136] f16 (rows 10..15 = 0)     ->    100,352 B
//  awbuf  : [4 chunk][16 m][16 f16] A-layout wts ->      2,048 B
//  gbuf   : cw[8], sw[8] f32                     ->         64 B
#define QF_BYTES    (size_t)(BATCH * 4 * 256 * 2)
#define FUSED_BYTES (size_t)(BATCH * KFC * 2)
#define FCW_BYTES   (size_t)(16 * KFC * 2)
#define AW_BYTES    (size_t)(4 * 16 * 16 * 2)

// ---------------------------------------------------------------- zero qf_pad
__global__ void k_zero(uint32_t* __restrict__ p) {
  p[blockIdx.x * blockDim.x + threadIdx.x] = 0u;   // exact-size launch
}

// ------------------------------------------------- fc_w f32 -> f16, pad N->16
__global__ void k_prep_fcw(const float* __restrict__ fc_w, _Float16* __restrict__ fcw16) {
  int t = blockIdx.x * blockDim.x + threadIdx.x;   // 0 .. 16*3136-1
  int n = t / KFC, k = t % KFC;
  fcw16[t] = (n < NCLS) ? (_Float16)fc_w[n * KFC + k] : (_Float16)0.f;
}

// ---- conv weights -> f16 in exact WMMA A lane order, permuted-padded K space
// K-slot = ci*16 + j ; j in [0,9) -> (ky,kx)=(j/3,j%3) ; j in [9,16) -> 0 pad.
// Buffer order: [cblk = kc*2+hi][m][i], where a[i] holds
//   K = kc*32 + 16*hi + i
__global__ void k_prep_convw(const float* __restrict__ conv_w, _Float16* __restrict__ aw) {
  int t = blockIdx.x * blockDim.x + threadIdx.x;   // 0..1023
  int i    = t & 15;
  int m    = (t >> 4) & 15;
  int hi   = (t >> 8) & 1;
  int kc   = (t >> 9) & 1;
  int K    = kc * 32 + 16 * hi + i;
  int ci   = K >> 4;
  int j    = K & 15;
  _Float16 v = (_Float16)0.f;
  if (j < 9) v = (_Float16)conv_w[m * 36 + ci * 9 + j];
  aw[t] = v;
}

// ------------------------------------------ gate angles: cw[8], sw[8] in f32
__global__ void k_prep_gates(const float* __restrict__ qw, float* __restrict__ g) {
  int t = threadIdx.x;
  if (t < 8) {
    float h = 0.5f * qw[t];
    g[t]     = cosf(h);
    g[8 + t] = sinf(h);
  }
}

// ------------------------------- patch conv + 4-qubit circuit, one thread/patch
__global__ void k_patch_quantum(const float* __restrict__ x,
                                const float* __restrict__ pw,
                                const float* __restrict__ pb,
                                const float* __restrict__ g,     // cw[8], sw[8]
                                _Float16* __restrict__ qf_pad) {
  int t = blockIdx.x * blockDim.x + threadIdx.x;   // exact: NPATCH threads
  int b   = t / NPIX;
  int pix = t % NPIX;
  int py = pix / 14, px = pix % 14;

  const float* xb = x + (size_t)b * 784 + (2 * py) * 28 + 2 * px;
  float x00 = xb[0], x01 = xb[1], x10 = xb[28], x11 = xb[29];

  float vc[4], vs[4];
#pragma unroll
  for (int c = 0; c < 4; ++c) {
    float p = pb[c] + x00 * pw[c * 4 + 0] + x01 * pw[c * 4 + 1]
                    + x10 * pw[c * 4 + 2] + x11 * pw[c * 4 + 3];
    float h = 0.5f * p;
    vc[c] = cosf(h); vs[c] = sinf(h);
  }

  // state[s], s = a*8+b*4+c*2+d ; wire w occupies bit (3-w)
  float st[16];
#pragma unroll
  for (int s = 0; s < 16; ++s) {
    float v = (s & 8) ? vs[0] : vc[0];
    v *= (s & 4) ? vs[1] : vc[1];
    v *= (s & 2) ? vs[2] : vc[2];
    v *= (s & 1) ? vs[3] : vc[3];
    st[s] = v;
  }

  float cw[8], sw[8];
#pragma unroll
  for (int i = 0; i < 8; ++i) { cw[i] = g[i]; sw[i] = g[8 + i]; }

#pragma unroll
  for (int i = 0; i < 8; ++i) {
    const int w  = i & 3, wt = (i + 1) & 3;
    const int mw = 8 >> w, mt = 8 >> wt;
    // RY on wire w
#pragma unroll
    for (int s = 0; s < 16; ++s) {
      if (!(s & mw)) {
        float a0 = st[s], a1 = st[s | mw];
        st[s]      = cw[i] * a0 - sw[i] * a1;
        st[s | mw] = sw[i] * a0 + cw[i] * a1;
      }
    }
    // CNOT(ctrl=w, tgt=wt): swap target bit where ctrl==1
#pragma unroll
    for (int s = 0; s < 16; ++s) {
      if ((s & mw) && !(s & mt)) {
        float tmp = st[s]; st[s] = st[s | mt]; st[s | mt] = tmp;
      }
    }
  }

  // Z expectations -> padded f16 frame at (py+1, px+1)
#pragma unroll
  for (int w = 0; w < 4; ++w) {
    const int mw = 8 >> w;
    float z = 0.f;
#pragma unroll
    for (int s = 0; s < 16; ++s) z += ((s & mw) ? -1.f : 1.f) * st[s] * st[s];
    qf_pad[((size_t)b * 4 + w) * 256 + (py + 1) * 16 + (px + 1)] = (_Float16)z;
  }
}

// ---------------- 3x3 conv 4->16 as implicit GEMM, one wave = 16 pixel columns
// K space permuted as in k_prep_convw: lane's 16-consecutive-K B fragment is
// exactly one input channel ci = kc*2 + hi; (ky,kx) are compile-time.
__global__ void k_conv3_wmma(const _Float16* __restrict__ qf_pad,
                             const _Float16* __restrict__ aw,     // [4][16][16]
                             const float* __restrict__ conv_b,    // [16]
                             _Float16* __restrict__ fused) {      // [B][3136]
  const int wave = (blockIdx.x * blockDim.x + threadIdx.x) >> 5;
  const int lane = threadIdx.x & 31;
  const int hi   = lane >> 4;
  const int nl   = lane & 15;

  const int q   = wave * 16 + nl;       // this lane's output pixel column
  const int b   = q / NPIX;
  const int pix = q % NPIX;
  const int py  = pix / 14, px = pix % 14;
  const _Float16* qp = qf_pad + (size_t)b * (4 * 256) + py * 16 + px;

  v8f acc = {};
#pragma unroll
  for (int kc = 0; kc < 2; ++kc) {
    const int cblk = kc * 2 + hi;       // input channel handled by this lane
    // A: one aligned 32B vector load in exact lane layout
    v16h a = *(const v16h*)(aw + ((cblk * 16 + nl) << 4));
    // B: 9 unconditional f16 loads (3x3 window in padded frame), rest zero
    const _Float16* qc = qp + cblk * 256;
    v16h bm = {};
#pragma unroll
    for (int j = 0; j < 9; ++j) {
      const int ky = j / 3, kx = j % 3;     // compile-time constants
      bm[j] = qc[ky * 16 + kx];
    }
    acc = __builtin_amdgcn_wmma_f32_16x16x32_f16(false, a, false, bm,
                                                 (short)0, acc, false, false);
  }
  // D: VGPR r -> out channel r+8*hi, column = pixel q ; add bias, store f16
#pragma unroll
  for (int r = 0; r < 8; ++r) {
    int mo = r + 8 * hi;
    float v = acc[r] + conv_b[mo];
    fused[(size_t)b * KFC + mo * NPIX + pix] = (_Float16)v;
  }
}

// -------- FC (16 batch rows x 16 cols, K=3136) + log_softmax, one wave = tile
__global__ void k_fc_softmax_wmma(const _Float16* __restrict__ fused, // [B][3136]
                                  const _Float16* __restrict__ fcw,   // [16][3136]
                                  const float* __restrict__ fc_b,     // [10]
                                  float* __restrict__ out) {          // [B][10]
  const int wave = (blockIdx.x * blockDim.x + threadIdx.x) >> 5;
  const int lane = threadIdx.x & 31;
  const int hi   = lane >> 4;
  const int nl   = lane & 15;
  const int b0   = wave * 16;

  const _Float16* arow = fused + (size_t)(b0 + nl) * KFC;  // A row = batch b0+nl
  const _Float16* brow = fcw   + (size_t)nl * KFC;         // B col = class nl

  v8f acc = {};
  for (int kc = 0; kc < 98; ++kc) {
    const int kbase = kc * 32;
    v16h a;
    a = *(const v16h*)(arow + kbase + 16 * hi);                 // K = kbase+16hi..+15
    v16h bm = *(const v16h*)(brow + kbase + 16 * hi);           // K = kbase+16hi..+15
    acc = __builtin_amdgcn_wmma_f32_16x16x32_f16(false, a, false, bm,
                                                 (short)0, acc, false, false);
  }

  // Epilogue: bias + row-wise log_softmax over the 10 valid columns.
  // Row m = r + 8*hi lives across the 16 lanes of this half-wave.
  const float bias = (nl < NCLS) ? fc_b[nl] : 0.f;
#pragma unroll
  for (int r = 0; r < 8; ++r) {
    const int m = r + 8 * hi;
    float val = acc[r] + bias;
    float mx = (nl < NCLS) ? val : -3.4e38f;
#pragma unroll
    for (int off = 8; off >= 1; off >>= 1) mx = fmaxf(mx, __shfl_xor(mx, off, 16));
    float e = (nl < NCLS) ? __expf(val - mx) : 0.f;
    float sm = e;
#pragma unroll
    for (int off = 8; off >= 1; off >>= 1) sm += __shfl_xor(sm, off, 16);
    if (nl < NCLS)
      out[(size_t)(b0 + m) * NCLS + nl] = val - mx - __logf(sm);
  }
}

extern "C" void kernel_launch(void* const* d_in, const int* in_sizes, int n_in,
                              void* d_out, int out_size, void* d_ws, size_t ws_size,
                              hipStream_t stream) {
  const float* x       = (const float*)d_in[0];
  const float* patch_w = (const float*)d_in[1];
  const float* patch_b = (const float*)d_in[2];
  const float* q_w     = (const float*)d_in[3];
  const float* conv_w  = (const float*)d_in[4];
  const float* conv_b  = (const float*)d_in[5];
  const float* fc_w    = (const float*)d_in[6];
  const float* fc_b    = (const float*)d_in[7];
  float* out = (float*)d_out;

  char* ws = (char*)d_ws;
  _Float16* qf_pad = (_Float16*)(ws);
  _Float16* fused  = (_Float16*)(ws + QF_BYTES);
  _Float16* fcw16  = (_Float16*)(ws + QF_BYTES + FUSED_BYTES);
  _Float16* awbuf  = (_Float16*)(ws + QF_BYTES + FUSED_BYTES + FCW_BYTES);
  float*    gbuf   = (float*)   (ws + QF_BYTES + FUSED_BYTES + FCW_BYTES + AW_BYTES);

  // 1) zero padded qf frame (borders must be 0 every call)
  k_zero<<<(QF_BYTES / 4) / 256, 256, 0, stream>>>((uint32_t*)qf_pad);
  // 2) weight prep: fc -> f16 (N padded), conv -> WMMA A layout, gate sincos
  k_prep_fcw<<<(16 * KFC) / 256, 256, 0, stream>>>(fc_w, fcw16);
  k_prep_convw<<<4, 256, 0, stream>>>(conv_w, awbuf);
  k_prep_gates<<<1, 32, 0, stream>>>(q_w, gbuf);
  // 3) patch conv + quantum circuit  (1,605,632 threads, exact)
  k_patch_quantum<<<NPATCH / 256, 256, 0, stream>>>(x, patch_w, patch_b, gbuf, qf_pad);
  // 4) 3x3 conv as WMMA implicit GEMM: 100,352 full waves
  k_conv3_wmma<<<(NPATCH / 16) * 32 / 256, 256, 0, stream>>>(qf_pad, awbuf, conv_b, fused);
  // 5) FC + log_softmax: 512 full waves
  k_fc_softmax_wmma<<<(BATCH / 16) * 32 / 256, 256, 0, stream>>>(fused, fcw16, fc_b, out);
}
